// SmolLM2Attention_19739669692721
// MI455X (gfx1250) — matrix operand order repaired and checked by reading, not compiled, not run on hardware
//
#include <hip/hip_runtime.h>
#include <hip/hip_bf16.h>

// ---------------------------------------------------------------------------
// SmolLM2 GQA attention forward for gfx1250 (MI455X), bf16 WMMA everywhere.
// B=4, S=2048, H=576, NH=9, NKV=3, HD=64, causal, RoPE theta=1e5.
// Async global->LDS staging (global_load_async_to_lds_b128 + s_wait_asynccnt)
// with guarded fallback to regular b128 load/ds_store copies.
// ---------------------------------------------------------------------------

typedef __bf16 bf16;
typedef __attribute__((ext_vector_type(16))) __bf16 v16bf;
typedef __attribute__((ext_vector_type(8)))  __bf16 v8bf;
typedef __attribute__((ext_vector_type(8)))  float  v8f;
typedef __attribute__((ext_vector_type(4)))  int    v4i_t;

#define Bb 4
#define Ss 2048
#define Hh 576
#define NHq 9
#define NKVq 3
#define HDd 64
#define NEGBIG (-1.0e9f)

#if __has_builtin(__builtin_amdgcn_global_load_async_to_lds_b128) && \
    __has_builtin(__builtin_amdgcn_s_wait_asynccnt)
#define HAVE_ASYNC_LDS 1
#else
#define HAVE_ASYNC_LDS 0
#endif

#if HAVE_ASYNC_LDS
// 16-byte async global->LDS copy (per-lane), tracked by ASYNCcnt.
__device__ __forceinline__ void async_copy16(void* lds, const void* g) {
  __builtin_amdgcn_global_load_async_to_lds_b128((v4i_t*)g, (v4i_t*)lds, 0, 0);
}
__device__ __forceinline__ void async_wait0() {
  __builtin_amdgcn_s_wait_asynccnt(0);
}
#else
__device__ __forceinline__ void async_copy16(void* lds, const void* g) {
  *(uint4*)lds = *(const uint4*)g;
}
__device__ __forceinline__ void async_wait0() {}
#endif

__device__ __forceinline__ v16bf cat16(v8bf lo, v8bf hi) {
  return __builtin_shufflevector(lo, hi, 0,1,2,3,4,5,6,7,8,9,10,11,12,13,14,15);
}
__device__ __forceinline__ v8f zero8() {
  v8f z;
#pragma unroll
  for (int i = 0; i < 8; ++i) z[i] = 0.0f;
  return z;
}

// ---------------------------------------------------------------------------
// fp32 -> bf16 elementwise
// ---------------------------------------------------------------------------
__global__ void cvt_x_kernel(const float* __restrict__ x, bf16* __restrict__ xb, int n) {
  int i = blockIdx.x * blockDim.x + threadIdx.x;
  if (i < n) xb[i] = (bf16)x[i];
}

// Pack [Wq;Wk;Wv] (960 x 576) and Wo (576 x 576) to bf16.
__global__ void pack_weights_kernel(const float* __restrict__ Wq, const float* __restrict__ Wk,
                                    const float* __restrict__ Wv, const float* __restrict__ Wo,
                                    bf16* __restrict__ wqkv, bf16* __restrict__ wo) {
  int idx = blockIdx.x * blockDim.x + threadIdx.x;
  const int NQKV = 960 * Hh;
  const int NWO  = Hh * Hh;
  if (idx < NQKV) {
    int n = idx / Hh, k = idx % Hh;
    float v;
    if (n < 576)      v = Wq[n * Hh + k];
    else if (n < 768) v = Wk[(n - 576) * Hh + k];
    else              v = Wv[(n - 768) * Hh + k];
    wqkv[idx] = (bf16)v;
  } else if (idx < NQKV + NWO) {
    int j = idx - NQKV;
    wo[j] = (bf16)Wo[j];
  }
}

// ---------------------------------------------------------------------------
// C[M,N] (f32) = A[M,K] (bf16) @ W[N,K]^T (bf16), WMMA bf16 16x16x32.
// Block: 128 threads (4 waves), tile 64(M) x 64(N), K-step 32.
// ---------------------------------------------------------------------------
__global__ __launch_bounds__(128) void gemm_bf16_kernel(
    const bf16* __restrict__ A, const bf16* __restrict__ W, float* __restrict__ C,
    int M, int N, int K) {
  __shared__ bf16 At[64 * 32];
  __shared__ bf16 Bt[64 * 32];

  const int tid  = threadIdx.x;
  const int lane = tid & 31;
  const int wave = tid >> 5;
  const int m0 = blockIdx.y * 64;
  const int n0 = blockIdx.x * 64;
  const int n16 = lane & 15;
  const int hi8 = (lane >> 4) * 8;     // D-frag row base per half-wave
  const int ko16 = (lane >> 4) * 16;   // B-frag K base per half-wave

  v8f acc[4];
#pragma unroll
  for (int t = 0; t < 4; ++t) acc[t] = zero8();

  // per-thread staging indices: 16 contiguous bf16 per thread per tile
  const int sidx = tid * 16;
  const int srow = sidx >> 5;
  const int scol = sidx & 31;

  for (int k0 = 0; k0 < K; k0 += 32) {
    __syncthreads();
    {
      const bf16* srcA = A + (size_t)(m0 + srow) * K + k0 + scol;
      async_copy16(&At[srow * 32 + scol],     srcA);
      async_copy16(&At[srow * 32 + scol + 8], srcA + 8);
      const bf16* srcB = W + (size_t)(n0 + srow) * K + k0 + scol;
      async_copy16(&Bt[srow * 32 + scol],     srcB);
      async_copy16(&Bt[srow * 32 + scol + 8], srcB + 8);
    }
    async_wait0();
    __syncthreads();

    const int arow = wave * 16 + n16;
    v8bf alo = *(const v8bf*)&At[arow * 32 + ko16];
    v8bf ahi = *(const v8bf*)&At[arow * 32 + ko16 + 8];
    v16bf afrag = cat16(alo, ahi);

#pragma unroll
    for (int t = 0; t < 4; ++t) {
      v16bf bfrag = *(const v16bf*)&Bt[(t * 16 + n16) * 32 + ko16];
      acc[t] = __builtin_amdgcn_wmma_f32_16x16x32_bf16(
          false, afrag, false, bfrag, (short)0, acc[t], false, false);
    }
  }

  float* Crow = C + (size_t)(m0 + wave * 16) * N + n0;
#pragma unroll
  for (int t = 0; t < 4; ++t)
#pragma unroll
    for (int r = 0; r < 8; ++r)
      Crow[(size_t)(r + hi8) * N + t * 16 + n16] = acc[t][r];
}

// ---------------------------------------------------------------------------
// RoPE on Q and K heads of the fp32 QKV buffer; scatter to per-head bf16.
// Q is pre-scaled by 1/sqrt(HD)=0.125. One thread per (b,s,head,freq-pair).
// ---------------------------------------------------------------------------
__global__ void rope_scatter_qk_kernel(const float* __restrict__ QKV,
                                       bf16* __restrict__ Qh, bf16* __restrict__ Kh) {
  int idx = blockIdx.x * blockDim.x + threadIdx.x;   // B*S*12*32
  const int i  = idx & 31;
  const int hh = (idx >> 5) % 12;        // 0..8 q-heads, 9..11 k-heads
  const int bs = idx / (12 * 32);
  const int s = bs % Ss;
  const int b = bs / Ss;
  const float* row = QKV + (size_t)bs * 960;
  const bool isQ = hh < NHq;
  const int col = isQ ? hh * HDd : Hh + (hh - NHq) * HDd;
  float x1 = row[col + i];
  float x2 = row[col + i + 32];
  float inv = powf(100000.0f, -(float)i * (1.0f / 32.0f));
  float ang = (float)s * inv;
  float c = cosf(ang), sn = sinf(ang);
  float o1 = x1 * c - x2 * sn;
  float o2 = x2 * c + x1 * sn;
  if (isQ) {
    o1 *= 0.125f; o2 *= 0.125f;
    size_t base = (((size_t)b * NHq + hh) * Ss + s) * HDd;
    Qh[base + i] = (bf16)o1; Qh[base + i + 32] = (bf16)o2;
  } else {
    int kv = hh - NHq;
    size_t base = (((size_t)b * NKVq + kv) * Ss + s) * HDd;
    Kh[base + i] = (bf16)o1; Kh[base + i + 32] = (bf16)o2;
  }
}

// V scattered d-major: Vt[b][kv][d][s] (coalesced writes along s).
__global__ void scatter_v_kernel(const float* __restrict__ QKV, bf16* __restrict__ Vt) {
  int idx = blockIdx.x * blockDim.x + threadIdx.x;   // B*NKV*HD*S
  const int s = idx & (Ss - 1);
  int rest = idx >> 11;
  const int d = rest & 63; rest >>= 6;
  const int kv = rest % NKVq;
  const int b  = rest / NKVq;
  Vt[idx] = (bf16)QKV[((size_t)(b * Ss + s)) * 960 + 768 + kv * HDd + d];
}

// ---------------------------------------------------------------------------
// Flash attention forward. Block = (q-tile of 64, head, batch), 128 threads.
// Wave w owns 16 q-rows; online softmax; WMMA for QK^T and PV.
// ---------------------------------------------------------------------------
__global__ __launch_bounds__(128) void attn_fwd_kernel(
    const bf16* __restrict__ Qh, const bf16* __restrict__ Kh,
    const bf16* __restrict__ Vt, bf16* __restrict__ O) {
  __shared__ bf16 Kt[64 * 64];        // [key][d]
  __shared__ bf16 Vtile[64 * 64];     // [d][key]
  __shared__ bf16 Pt[4 * 16 * 64];    // per-wave P tiles [16 q][64 key]

  const int tid  = threadIdx.x;
  const int lane = tid & 31;
  const int wave = tid >> 5;
  const int qt = blockIdx.x, h = blockIdx.y, b = blockIdx.z;
  const int kv = h / (NHq / NKVq);
  const int q0 = qt * 64;
  const int n16 = lane & 15;
  const int hi8 = (lane >> 4) * 8;
  const int ko16 = (lane >> 4) * 16;

  // Q fragments (A-layout), 2 d-chunks of 32, live for whole kernel.
  const int qrow = q0 + wave * 16 + n16;
  const size_t qbase = (((size_t)b * NHq + h) * Ss + qrow) * HDd;
  v16bf qfrag[2];
#pragma unroll
  for (int c = 0; c < 2; ++c) {
    v8bf lo  = *(const v8bf*)(Qh + qbase + c * 32 + ko16);
    v8bf hiv = *(const v8bf*)(Qh + qbase + c * 32 + ko16 + 8);
    qfrag[c] = cat16(lo, hiv);
  }

  const size_t kbase = (((size_t)b * NKVq + kv) * Ss) * HDd;   // Kh head base
  const size_t vbase = (((size_t)b * NKVq + kv) * HDd) * Ss;   // Vt head base

  v8f acc[4];
#pragma unroll
  for (int t = 0; t < 4; ++t) acc[t] = zero8();
  float mrow[8], lrow[8];
#pragma unroll
  for (int r = 0; r < 8; ++r) { mrow[r] = -3.0e38f; lrow[r] = 0.0f; }

  bf16* Pw = Pt + wave * 16 * 64;
  const int nTiles = qt + 1;                 // causal: keys <= q0+63
  const int qgbase = q0 + wave * 16;

  for (int jt = 0; jt < nTiles; ++jt) {
    const int j = jt * 64;
    __syncthreads();
    // Stage K tile [64 key][64 d] and V tile [64 d][64 key] via async
    // global->LDS b128 copies (ASYNCcnt), 16 B per lane per copy.
#pragma unroll
    for (int it = 0; it < 4; ++it) {
      int e = (tid + it * 128) * 8;
      int row = e >> 6, col = e & 63;
      async_copy16(&Kt[e],    Kh + kbase + (size_t)(j + row) * HDd + col);
      async_copy16(&Vtile[e], Vt + vbase + (size_t)row * Ss + j + col);
    }
    async_wait0();
    __syncthreads();

    if (jt + 1 < nTiles) {   // gfx1250 global_prefetch_b8 on next tiles
      __builtin_prefetch(Kh + kbase + (size_t)(j + 64) * HDd + tid * 32, 0, 1);
      __builtin_prefetch(Vt + vbase + (size_t)(tid & 63) * Ss + j + 64, 0, 1);
    }

    // ---- scores: 4 tiles of 16q x 16k, K-dim = 64 (2 WMMA each) ----
    v8f st[4];
#pragma unroll
    for (int tk = 0; tk < 4; ++tk) {
      v16bf bk0 = *(const v16bf*)&Kt[(tk * 16 + n16) * 64 + ko16];
      v16bf bk1 = *(const v16bf*)&Kt[(tk * 16 + n16) * 64 + 32 + ko16];
      v8f s0 = zero8();
      s0 = __builtin_amdgcn_wmma_f32_16x16x32_bf16(false, qfrag[0], false, bk0, (short)0, s0, false, false);
      s0 = __builtin_amdgcn_wmma_f32_16x16x32_bf16(false, qfrag[1], false, bk1, (short)0, s0, false, false);
      st[tk] = s0;
    }

    // ---- causal mask (D layout: row = r+hi8, col = tk*16+n16) ----
#pragma unroll
    for (int tk = 0; tk < 4; ++tk) {
      int key = j + tk * 16 + n16;
#pragma unroll
      for (int r = 0; r < 8; ++r) {
        int qi = qgbase + r + hi8;
        if (key > qi) st[tk][r] = NEGBIG;
      }
    }

    // ---- online softmax ----
#pragma unroll
    for (int r = 0; r < 8; ++r) {
      float mx = fmaxf(fmaxf(st[0][r], st[1][r]), fmaxf(st[2][r], st[3][r]));
#pragma unroll
      for (int o = 1; o < 16; o <<= 1) mx = fmaxf(mx, __shfl_xor(mx, o, 32));
      float mnew = fmaxf(mrow[r], mx);
      float alpha = __expf(mrow[r] - mnew);
      mrow[r] = mnew;
      float sum = 0.0f;
#pragma unroll
      for (int tk = 0; tk < 4; ++tk) {
        float p = __expf(st[tk][r] - mnew);
        st[tk][r] = p;
        sum += p;
      }
#pragma unroll
      for (int o = 1; o < 16; o <<= 1) sum += __shfl_xor(sum, o, 32);
      lrow[r] = lrow[r] * alpha + sum;
#pragma unroll
      for (int to = 0; to < 4; ++to) acc[to][r] *= alpha;
    }

    // ---- D-layout -> A-layout via per-wave LDS P tile ----
#pragma unroll
    for (int tk = 0; tk < 4; ++tk)
#pragma unroll
      for (int r = 0; r < 8; ++r)
        Pw[(r + hi8) * 64 + tk * 16 + n16] = (bf16)st[tk][r];

    // ---- O += P @ V  (K-dim = 64 keys in 2 chunks, 4 d-subtiles) ----
#pragma unroll
    for (int kc = 0; kc < 2; ++kc) {
      v8bf plo = *(const v8bf*)&Pw[n16 * 64 + kc * 32 + ko16];
      v8bf phi = *(const v8bf*)&Pw[n16 * 64 + kc * 32 + ko16 + 8];
      v16bf pfrag = cat16(plo, phi);
#pragma unroll
      for (int to = 0; to < 4; ++to) {
        v16bf bv = *(const v16bf*)&Vtile[(to * 16 + n16) * 64 + kc * 32 + ko16];
        acc[to] = __builtin_amdgcn_wmma_f32_16x16x32_bf16(
            false, pfrag, false, bv, (short)0, acc[to], false, false);
      }
    }
  }

  // ---- normalize and store O[b][s][h*64+d] (bf16, GEMM-A layout) ----
#pragma unroll
  for (int r = 0; r < 8; ++r) {
    float inv = 1.0f / lrow[r];
#pragma unroll
    for (int to = 0; to < 4; ++to) acc[to][r] *= inv;
  }
#pragma unroll
  for (int r = 0; r < 8; ++r) {
    size_t orow = ((size_t)b * Ss + q0 + wave * 16 + r + hi8) * Hh + h * HDd;
#pragma unroll
    for (int to = 0; to < 4; ++to)
      O[orow + to * 16 + n16] = (bf16)acc[to][r];
  }
}

// ---------------------------------------------------------------------------
extern "C" void kernel_launch(void* const* d_in, const int* in_sizes, int n_in,
                              void* d_out, int out_size, void* d_ws, size_t ws_size,
                              hipStream_t stream) {
  const float* x  = (const float*)d_in[0];
  const float* Wq = (const float*)d_in[1];
  const float* Wk = (const float*)d_in[2];
  const float* Wv = (const float*)d_in[3];
  const float* Wo = (const float*)d_in[4];
  // d_in[5] = mask (unused; causal mask computed analytically)

  const int BS = Bb * Ss;                  // 8192 rows
  char* p = (char*)d_ws;
  bf16* xb    = (bf16*)p; p += (size_t)BS * Hh * 2;            // 9,437,184
  bf16* wqkv  = (bf16*)p; p += (size_t)960 * Hh * 2;           // 1,105,920
  bf16* wo_b  = (bf16*)p; p += (size_t)Hh * Hh * 2;            //   663,552
  float* qkvf = (float*)p; p += (size_t)BS * 960 * 4;          // 31,457,280
  bf16* Obuf  = (bf16*)qkvf;   // alias: QKV fp32 is dead once attn runs
  bf16* Qh    = (bf16*)p; p += (size_t)Bb * NHq * Ss * HDd * 2;  // 9,437,184
  bf16* Kh    = (bf16*)p; p += (size_t)Bb * NKVq * Ss * HDd * 2; // 3,145,728
  bf16* Vt    = (bf16*)p; p += (size_t)Bb * NKVq * HDd * Ss * 2; // 3,145,728

  // 1) fp32 -> bf16 conversions / weight packing
  cvt_x_kernel<<<(BS * Hh) / 256, 256, 0, stream>>>(x, xb, BS * Hh);
  {
    int nw = 960 * Hh + Hh * Hh;
    pack_weights_kernel<<<(nw + 255) / 256, 256, 0, stream>>>(Wq, Wk, Wv, Wo, wqkv, wo_b);
  }
  // 2) fused QKV projection: [8192 x 960] = xb @ wqkv^T
  gemm_bf16_kernel<<<dim3(960 / 64, BS / 64), 128, 0, stream>>>(xb, wqkv, qkvf, BS, 960, Hh);
  // 3) RoPE + per-head scatter (Q scaled by 1/8), V scattered d-major
  rope_scatter_qk_kernel<<<(Bb * Ss * 12 * 32) / 256, 256, 0, stream>>>(qkvf, Qh, Kh);
  scatter_v_kernel<<<(Bb * NKVq * HDd * Ss) / 256, 256, 0, stream>>>(qkvf, Vt);
  // 4) causal flash attention
  attn_fwd_kernel<<<dim3(Ss / 64, NHq, Bb), 128, 0, stream>>>(Qh, Kh, Vt, Obuf);
  // 5) output projection: d_out[8192 x 576] = Obuf @ Wo^T (fp32 out)
  gemm_bf16_kernel<<<dim3(Hh / 64, BS / 64), 128, 0, stream>>>(Obuf, wo_b, (float*)d_out, BS, Hh, Hh);
}
